// LinearSparseAttention_57002805952725
// MI455X (gfx1250) — matrix operand order repaired and checked by reading, not compiled, not run on hardware
//
#include <hip/hip_runtime.h>
#include <stdint.h>

typedef unsigned short u16;
typedef __attribute__((ext_vector_type(16))) __bf16 v16bf;
typedef __attribute__((ext_vector_type(16))) u16    v16u;
typedef __attribute__((ext_vector_type(8)))  float  v8f;

#define B_  4
#define S_  8192
#define D_  1024
#define H_  16
#define HD_ 64
#define W_  512
#define NB_ 16
#define M_  (B_ * S_)

// ---------- helpers ----------

__device__ __forceinline__ u16 f2bf(float f) {
    unsigned int u = __builtin_bit_cast(unsigned int, f);
    u += 0x7FFFu + ((u >> 16) & 1u);   // round-to-nearest-even
    return (u16)(u >> 16);
}

__device__ __forceinline__ unsigned int pk_bf16(float a, float b) {
#if __has_builtin(__builtin_amdgcn_cvt_pk_bf16_f32)
    auto t = __builtin_amdgcn_cvt_pk_bf16_f32(a, b);
    return __builtin_bit_cast(unsigned int, t);
#else
    return (unsigned int)f2bf(a) | ((unsigned int)f2bf(b) << 16);
#endif
}

__device__ __forceinline__ float fast_rcp(float x) {
#if __has_builtin(__builtin_amdgcn_rcpf)
    return __builtin_amdgcn_rcpf(x);
#else
    return 1.0f / x;
#endif
}

__device__ __forceinline__ v8f vzero8() {
    v8f z;
#pragma unroll
    for (int i = 0; i < 8; ++i) z[i] = 0.0f;
    return z;
}

__device__ __forceinline__ v8f wmma_bf16(v16u a, v16u b, v8f c) {
    // D = A(16x32 bf16) * B(32x16 bf16) + C(16x16 f32)
    return __builtin_amdgcn_wmma_f32_16x16x32_bf16(
        /*neg_a=*/false, __builtin_bit_cast(v16bf, a),
        /*neg_b=*/false, __builtin_bit_cast(v16bf, b),
        /*c_mod=*/(short)0, c, /*reuse_a=*/false, /*reuse_b=*/false);
}

// max-reduce across each 16-lane group (butterfly xor 1,2,4,8)
__device__ __forceinline__ float rowmax16(float x) {
#if __has_builtin(__builtin_amdgcn_permlane16)
    unsigned int u, t;
    u = __builtin_bit_cast(unsigned int, x);
    t = __builtin_amdgcn_permlane16(u, u, 0x67452301u, 0xEFCDAB89u, false, false); // xor 1
    x = fmaxf(x, __builtin_bit_cast(float, t));
    u = __builtin_bit_cast(unsigned int, x);
    t = __builtin_amdgcn_permlane16(u, u, 0x54761032u, 0xDCFE98BAu, false, false); // xor 2
    x = fmaxf(x, __builtin_bit_cast(float, t));
    u = __builtin_bit_cast(unsigned int, x);
    t = __builtin_amdgcn_permlane16(u, u, 0x32107654u, 0xBA98FEDCu, false, false); // xor 4
    x = fmaxf(x, __builtin_bit_cast(float, t));
    u = __builtin_bit_cast(unsigned int, x);
    t = __builtin_amdgcn_permlane16(u, u, 0xFEDCBA98u, 0x76543210u, false, false); // xor 8
    x = fmaxf(x, __builtin_bit_cast(float, t));
    return x;
#else
    x = fmaxf(x, __shfl_xor(x, 1, 16));
    x = fmaxf(x, __shfl_xor(x, 2, 16));
    x = fmaxf(x, __shfl_xor(x, 4, 16));
    x = fmaxf(x, __shfl_xor(x, 8, 16));
    return x;
#endif
}

// ---------- fp32 -> bf16 convert ----------

__global__ __launch_bounds__(256) void cvt_f32_bf16(const float* __restrict__ in,
                                                    u16* __restrict__ out,
                                                    long long n) {
    long long i = (long long)blockIdx.x * blockDim.x + threadIdx.x;
    long long stride = (long long)gridDim.x * blockDim.x;
    for (; i < n; i += stride) out[i] = f2bf(in[i]);
}

// ---------- bf16 GEMM:  C[M,N] = (A[M,K] @ W[N,K]^T + bias) * outScale ----------
// block = 256 threads (8 waves), tile 128x128, K-chunk 32.

template <bool OUT_F32>
__global__ __launch_bounds__(256) void gemm_bias_bf16(const u16* __restrict__ A,
                                                      const u16* __restrict__ Wt,
                                                      const float* __restrict__ bias,
                                                      void* __restrict__ Cout,
                                                      float outScale) {
    __shared__ __attribute__((aligned(32))) u16 At[128 * 32];
    __shared__ __attribute__((aligned(32))) u16 Bt[128 * 32];

    const int nTiles = D_ / 128;  // 8
    const int m0 = (blockIdx.x / nTiles) * 128;
    const int n0 = (blockIdx.x % nTiles) * 128;

    const int lane = threadIdx.x & 31;
    const int wave = threadIdx.x >> 5;
    const int l16 = lane & 15;
    const int half = lane >> 4;
    const int wm = wave >> 1;   // 0..3 : 32-row strip
    const int wn = wave & 1;    // 0..1 : 64-col strip

    v8f acc[2][4];
#pragma unroll
    for (int i = 0; i < 2; ++i)
#pragma unroll
        for (int j = 0; j < 4; ++j) acc[i][j] = vzero8();

    const int ldr = threadIdx.x >> 1;          // 0..127
    const int ldc = (threadIdx.x & 1) * 16;    // 0 or 16

    for (int k0 = 0; k0 < D_; k0 += 32) {
        *(v16u*)&At[ldr * 32 + ldc] = *(const v16u*)(A  + (size_t)(m0 + ldr) * D_ + k0 + ldc);
        *(v16u*)&Bt[ldr * 32 + ldc] = *(const v16u*)(Wt + (size_t)(n0 + ldr) * D_ + k0 + ldc);
        if (k0 + 32 < D_) {
            __builtin_prefetch(A  + (size_t)(m0 + ldr) * D_ + k0 + 32 + ldc, 0, 1);
            __builtin_prefetch(Wt + (size_t)(n0 + ldr) * D_ + k0 + 32 + ldc, 0, 1);
        }
        __syncthreads();

        v16u af[2], bf[4];
#pragma unroll
        for (int ms = 0; ms < 2; ++ms)
            af[ms] = *(const v16u*)&At[(wm * 32 + ms * 16 + l16) * 32 + half * 16];
#pragma unroll
        for (int ns = 0; ns < 4; ++ns)
            bf[ns] = *(const v16u*)&Bt[(wn * 64 + ns * 16 + l16) * 32 + half * 16];

#pragma unroll
        for (int ms = 0; ms < 2; ++ms)
#pragma unroll
            for (int ns = 0; ns < 4; ++ns)
                acc[ms][ns] = wmma_bf16(af[ms], bf[ns], acc[ms][ns]);

        __syncthreads();
    }

    // epilogue: D-layout = VGPR r -> row r + 8*half, col = l16
#pragma unroll
    for (int ms = 0; ms < 2; ++ms)
#pragma unroll
        for (int ns = 0; ns < 4; ++ns)
#pragma unroll
            for (int r = 0; r < 8; ++r) {
                int gm = m0 + wm * 32 + ms * 16 + r + half * 8;
                int gn = n0 + wn * 64 + ns * 16 + l16;
                float v = (acc[ms][ns][r] + bias[gn]) * outScale;
                if (OUT_F32)
                    ((float*)Cout)[(size_t)gm * D_ + gn] = v;
                else
                    ((u16*)Cout)[(size_t)gm * D_ + gn] = f2bf(v);
            }
}

// ---------- windowed flash attention ----------
// grid = B*NB*H*4 ; block = 256 threads (8 waves), each wave owns 16 query rows.
// Q pre-scaled by 1/sqrt(HD) in the Q projection.
// Keys: 1024 (= previous block + current block), 64 at a time from LDS.

__global__ __launch_bounds__(256) void attn_win_kernel(const u16* __restrict__ qb,
                                                       const u16* __restrict__ kb,
                                                       const u16* __restrict__ vb,
                                                       u16* __restrict__ ob) {
    __shared__ __attribute__((aligned(32))) u16 lds_k[64 * 64];        // [key][hd]
    __shared__ __attribute__((aligned(32))) u16 lds_vt[64 * 64];       // [hd][key]
    __shared__ __attribute__((aligned(32))) unsigned int lds_p[8][16 * 32]; // packed P per wave

    const int bid = blockIdx.x;
    const int qc = bid & 3;
    const int h  = (bid >> 2) & (H_ - 1);
    const int n  = (bid >> 6) & (NB_ - 1);
    const int b  = bid >> 10;

    const int lane = threadIdx.x & 31;
    const int wave = threadIdx.x >> 5;
    const int l16 = lane & 15;
    const int half = lane >> 4;

    const int qrow = n * W_ + qc * 128 + wave * 16;

    // Q fragments: 16 rows x 64 hd -> two A-fragments (K=0..31, 32..63)
    const u16* qptr = qb + ((size_t)(b * S_ + qrow + l16) * H_ + h) * HD_;
    const v16u aq0 = *(const v16u*)(qptr + half * 16);
    const v16u aq1 = *(const v16u*)(qptr + 32 + half * 16);

    // all-ones bf16 B fragment for WMMA row-sum
    v16u ones;
#pragma unroll
    for (int i = 0; i < 16; ++i) ones[i] = 0x3F80u;   // bf16(1.0)

    float mrow[8], lsum[8];
    v8f o[4];
#pragma unroll
    for (int r = 0; r < 8; ++r) { mrow[r] = -1e30f; lsum[r] = 0.0f; }
#pragma unroll
    for (int g = 0; g < 4; ++g) o[g] = vzero8();

    const int t0 = (n == 0) ? 8 : 0;    // exactly skips the -inf padded region
    const int keyBase = n * W_ - W_;

    const int cr = threadIdx.x >> 2;        // 0..63 : key row for cooperative loads
    const int cc = (threadIdx.x & 3) * 16;  // 0/16/32/48

    for (int t = t0; t < 16; ++t) {
        const int krow = keyBase + t * 64;

        // cooperative stage: K chunk (row-major), V chunk (transposed)
        const size_t kvoff = ((size_t)(b * S_ + krow + cr) * H_ + h) * HD_ + cc;
        *(v16u*)&lds_k[cr * 64 + cc] = *(const v16u*)(kb + kvoff);
        {
            const u16* vsrc = vb + kvoff;
#pragma unroll
            for (int e = 0; e < 16; ++e) lds_vt[(cc + e) * 64 + ((cr & 32) | (((cr >> 3) & 1) << 4) | ((cr & 7) << 1) | ((cr >> 4) & 1))] = vsrc[e];
        }
        if (t + 1 < 16) {
            __builtin_prefetch(kb + kvoff + (size_t)64 * H_ * HD_, 0, 1);
            __builtin_prefetch(vb + kvoff + (size_t)64 * H_ * HD_, 0, 1);
        }
        __syncthreads();

        // scores: four 16x16 tiles (key groups of 16), contraction over hd=64
        v8f s[4];
#pragma unroll
        for (int j = 0; j < 4; ++j) {
            const v16u bk_lo = *(const v16u*)&lds_k[(j * 16 + l16) * 64 + half * 16];
            const v16u bk_hi = *(const v16u*)&lds_k[(j * 16 + l16) * 64 + 32 + half * 16];
            s[j] = wmma_bf16(aq0, bk_lo, vzero8());
            s[j] = wmma_bf16(aq1, bk_hi, s[j]);
        }

        // online softmax: max via permlane16 butterfly, sum via WMMA-with-ones
        float p[4][8], sf[8];
#pragma unroll
        for (int r = 0; r < 8; ++r) {
            float mx = fmaxf(fmaxf(s[0][r], s[1][r]), fmaxf(s[2][r], s[3][r]));
            mx = rowmax16(mx);
            float mn = fmaxf(mrow[r], mx);
            sf[r] = __expf(mrow[r] - mn);
#pragma unroll
            for (int j = 0; j < 4; ++j) p[j][r] = __expf(s[j][r] - mn);
            mrow[r] = mn;
        }

        // pack P (16x64 bf16) into per-wave LDS staging: 16 x ds_store_b32
        unsigned int* pst = lds_p[wave];
#pragma unroll
        for (int r = 0; r < 8; ++r) {
            int row = r + half * 8;
            pst[row * 32 + l16]      = pk_bf16(p[0][r], p[1][r]);
            pst[row * 32 + 16 + l16] = pk_bf16(p[2][r], p[3][r]);
        }
        const v16u pa0 = *(const v16u*)&pst[l16 * 32 + half * 8];
        const v16u pa1 = *(const v16u*)&pst[l16 * 32 + 16 + half * 8];

        // row-sum of P, replicated across each 16-lane group (rows r+8*half)
        v8f rs = wmma_bf16(pa1, ones, wmma_bf16(pa0, ones, vzero8()));
#pragma unroll
        for (int r = 0; r < 8; ++r) {
            lsum[r] = lsum[r] * sf[r] + rs[r];
#pragma unroll
            for (int g = 0; g < 4; ++g) o[g][r] *= sf[r];
        }

        // O += P(16x64) @ V(64x64)
#pragma unroll
        for (int g = 0; g < 4; ++g) {
            const v16u bv0 = *(const v16u*)&lds_vt[(g * 16 + l16) * 64 + half * 16];
            const v16u bv1 = *(const v16u*)&lds_vt[(g * 16 + l16) * 64 + 32 + half * 16];
            o[g] = wmma_bf16(pa0, bv0, o[g]);
            o[g] = wmma_bf16(pa1, bv1, o[g]);
        }
        __syncthreads();
    }

    // normalize (rcp + mul) and store bf16
#pragma unroll
    for (int r = 0; r < 8; ++r) {
        float inv = fast_rcp(lsum[r]);
        int row = qrow + r + half * 8;
#pragma unroll
        for (int g = 0; g < 4; ++g) {
            int hd = g * 16 + l16;
            ob[((size_t)(b * S_ + row) * H_ + h) * HD_ + hd] = f2bf(o[g][r] * inv);
        }
    }
}

// ---------- launcher ----------

extern "C" void kernel_launch(void* const* d_in, const int* in_sizes, int n_in,
                              void* d_out, int out_size, void* d_ws, size_t ws_size,
                              hipStream_t stream) {
    const float* x  = (const float*)d_in[0];
    const float* wq = (const float*)d_in[1];
    const float* bq = (const float*)d_in[2];
    const float* wk = (const float*)d_in[3];
    const float* bk = (const float*)d_in[4];
    const float* wv = (const float*)d_in[5];
    const float* bv = (const float*)d_in[6];
    const float* wo = (const float*)d_in[7];
    const float* bo = (const float*)d_in[8];

    const size_t szX = (size_t)M_ * D_ * sizeof(u16);  // 64 MB
    const size_t szW = (size_t)D_ * D_ * sizeof(u16);  // 2 MB

    char* p = (char*)d_ws;
    u16* xb  = (u16*)p; p += szX;
    u16* wqb = (u16*)p; p += szW;
    u16* wkb = (u16*)p; p += szW;
    u16* wvb = (u16*)p; p += szW;
    u16* wob = (u16*)p; p += szW;
    u16* qb  = (u16*)p; p += szX;
    u16* kb_ = (u16*)p; p += szX;
    u16* vb_ = (u16*)p; p += szX;
    u16* ab  = (u16*)p; p += szX;

    // fp32 -> bf16
    cvt_f32_bf16<<<2048, 256, 0, stream>>>(x,  xb,  (long long)M_ * D_);
    cvt_f32_bf16<<<512,  256, 0, stream>>>(wq, wqb, (long long)D_ * D_);
    cvt_f32_bf16<<<512,  256, 0, stream>>>(wk, wkb, (long long)D_ * D_);
    cvt_f32_bf16<<<512,  256, 0, stream>>>(wv, wvb, (long long)D_ * D_);
    cvt_f32_bf16<<<512,  256, 0, stream>>>(wo, wob, (long long)D_ * D_);

    const int gemmGrid = (M_ / 128) * (D_ / 128);  // 2048

    // projections: fold 1/sqrt(HD)=0.125 into Q
    gemm_bias_bf16<false><<<gemmGrid, 256, 0, stream>>>(xb, wqb, bq, qb, 0.125f);
    gemm_bias_bf16<false><<<gemmGrid, 256, 0, stream>>>(xb, wkb, bk, kb_, 1.0f);
    gemm_bias_bf16<false><<<gemmGrid, 256, 0, stream>>>(xb, wvb, bv, vb_, 1.0f);

    // windowed attention
    attn_win_kernel<<<B_ * NB_ * H_ * 4, 256, 0, stream>>>(qb, kb_, vb_, ab);

    // output projection (fp32 out)
    gemm_bias_bf16<true><<<gemmGrid, 256, 0, stream>>>(ab, wob, bo, (float*)d_out, 1.0f);
}
